// TransformerBlock_63101659513133
// MI455X (gfx1250) — compile-verified, matrix operand order checked by reading, not run on hardware
//
#include <hip/hip_runtime.h>
#include <math.h>

typedef __attribute__((ext_vector_type(8)))  float        v8f;
typedef __attribute__((ext_vector_type(16))) __bf16       v16bf;
typedef __attribute__((ext_vector_type(4)))  unsigned int u32x4;
typedef __attribute__((ext_vector_type(4)))  float        f32x4;

#define D_MODELc 1024
#define N_HEADSc 16
#define D_HEADc  64
#define D_MLPc   4096
#define Bc       4
#define Sc       2048
#define MROWS    (Bc * Sc)   // 8192

// ---------------- helpers ----------------

__device__ __forceinline__ unsigned short f2bf(float f) {
    union { float f; unsigned u; } x; x.f = f;
    unsigned r = x.u + 0x7FFFu + ((x.u >> 16) & 1u);
    return (unsigned short)(r >> 16);
}

union FragAB { v16bf v; u32x4 q[2]; };

__device__ __forceinline__ void load_frag(FragAB& f, const unsigned short* p0,
                                          const unsigned short* p1) {
    f.q[0] = *reinterpret_cast<const u32x4*>(p0);
    f.q[1] = *reinterpret_cast<const u32x4*>(p1);
}

__device__ __forceinline__ v8f wmma_bf16(v16bf a, v16bf b, v8f c) {
    return __builtin_amdgcn_wmma_f32_16x16x32_bf16(false, a, false, b,
                                                   (short)0, c, false, false);
}

// gfx1250 async global->LDS copy (ASYNCcnt tracked, no VGPR round trip).
__device__ __forceinline__ void async_copy_b128(unsigned ldsoff, const unsigned short* g) {
    asm volatile("global_load_async_to_lds_b128 %0, %1, off"
                 :: "v"(ldsoff), "v"(g) : "memory");
}
__device__ __forceinline__ void wait_async0() {
    asm volatile("s_wait_asynccnt 0" ::: "memory");
}
__device__ __forceinline__ unsigned lds_offset(const void* p) {
    return (unsigned)(unsigned long long)p;   // LDS aperture lives in the high 32 bits
}

// ---------------- shared GEMM core ----------------
// acc[2][4] += A[m0..m0+31][0..K) * Bt[n0..n0+63][0..K)^T
// B tile (64x32 bf16 = 4KB) async-staged into LDS, double buffered.
// All 4 B fragments are pulled into distinct registers before the WMMA burst
// so the 8 ds_load_b128 issue back-to-back and wait once.
__device__ __forceinline__ void gemm_core(const unsigned short* __restrict__ A,
                                          const unsigned short* __restrict__ Bt,
                                          int K, int m0, int n0,
                                          v8f acc[2][4],
                                          unsigned short* lds /* [2][64][32] */) {
    const int tid  = threadIdx.x;
    const int lane = tid & 31;
    const int sel  = lane >> 4;
    const int l16  = lane & 15;

    // cooperative copy mapping: thread t -> (n = t/4, k-sub = (t%4)*8), one b128 each
    const int cn = tid >> 2;
    const int ck = (tid & 3) * 8;
    const unsigned short* gB = Bt + (size_t)(n0 + cn) * K + ck;
    const unsigned loff0 = lds_offset(lds + cn * 32 + ck);
    const unsigned loff1 = lds_offset(lds + 2048 + cn * 32 + ck);

    const unsigned short* arow0 = A + (size_t)(m0 + l16) * K;
    const unsigned short* arow1 = A + (size_t)(m0 + 16 + l16) * K;
    // per-wave LDS read base for B fragments (row n' = j*16 + l16)
    const unsigned short* lbase = lds + l16 * 32 + sel * 8;

    async_copy_b128(loff0, gB);   // prologue: tile 0 -> buffer 0

    for (int k = 0; k < K; k += 32) {
        const int buf = (k >> 5) & 1;
        wait_async0();        // my portion of tile k has landed
        __syncthreads();      // everyone's portion landed; buf^1 free to overwrite
        if (k + 32 < K)
            async_copy_b128(buf ? loff0 : loff1, gB + k + 32);

        FragAB a0, a1;
        load_frag(a0, arow0 + k + sel * 8, arow0 + k + 16 + sel * 8);
        load_frag(a1, arow1 + k + sel * 8, arow1 + k + 16 + sel * 8);

        const unsigned short* lb = lbase + buf * 2048;
        FragAB b0, b1, b2, b3;                 // distinct live values
        load_frag(b0, lb,              lb + 16);
        load_frag(b1, lb + 16 * 32,    lb + 16 * 32 + 16);
        load_frag(b2, lb + 32 * 32,    lb + 32 * 32 + 16);
        load_frag(b3, lb + 48 * 32,    lb + 48 * 32 + 16);

        acc[0][0] = wmma_bf16(a0.v, b0.v, acc[0][0]);
        acc[1][0] = wmma_bf16(a1.v, b0.v, acc[1][0]);
        acc[0][1] = wmma_bf16(a0.v, b1.v, acc[0][1]);
        acc[1][1] = wmma_bf16(a1.v, b1.v, acc[1][1]);
        acc[0][2] = wmma_bf16(a0.v, b2.v, acc[0][2]);
        acc[1][2] = wmma_bf16(a1.v, b2.v, acc[1][2]);
        acc[0][3] = wmma_bf16(a0.v, b3.v, acc[0][3]);
        acc[1][3] = wmma_bf16(a1.v, b3.v, acc[1][3]);
    }
}

// ---------------- weight transpose + f32->bf16 ----------------
__global__ void transpose_cvt_kernel(const float* __restrict__ src,
                                     unsigned short* __restrict__ dst,
                                     int R, int C,
                                     long srcBatchStride, long dstBatchStride,
                                     int dstRowStride) {
    long z   = blockIdx.z;
    long idx = (long)blockIdx.x * 256 + threadIdx.x;
    if (idx >= (long)R * C) return;
    int r = (int)(idx / C);
    int c = (int)(idx % C);
    dst[z * dstBatchStride + (long)c * dstRowStride + r] =
        f2bf(src[z * srcBatchStride + idx]);
}

// ---------------- layernorm -> bf16 ----------------
__global__ __launch_bounds__(256)
void ln_bf16_kernel(const float* __restrict__ x, const float* __restrict__ g,
                    const float* __restrict__ bt, unsigned short* __restrict__ out) {
    int row = blockIdx.x;
    int tid = threadIdx.x;
    const float* xr = x + (size_t)row * D_MODELc;
    f32x4 v = reinterpret_cast<const f32x4*>(xr)[tid];
    float s  = v[0] + v[1] + v[2] + v[3];
    float s2 = v[0]*v[0] + v[1]*v[1] + v[2]*v[2] + v[3]*v[3];
    __shared__ float sh1[256], sh2[256];
    sh1[tid] = s; sh2[tid] = s2;
    __syncthreads();
    for (int off = 128; off > 0; off >>= 1) {
        if (tid < off) { sh1[tid] += sh1[tid + off]; sh2[tid] += sh2[tid + off]; }
        __syncthreads();
    }
    float mu   = sh1[0] * (1.0f / D_MODELc);
    float var  = sh2[0] * (1.0f / D_MODELc) - mu * mu;
    float rstd = rsqrtf(var + 1e-5f);
    int c = tid * 4;
    unsigned short* orow = out + (size_t)row * D_MODELc + c;
#pragma unroll
    for (int i = 0; i < 4; ++i)
        orow[i] = f2bf((v[i] - mu) * rstd * g[c + i] + bt[c + i]);
}

// ---------------- generic GEMM: C = A(bf16,[M,K]) * Bt(bf16,[N,K])^T ----------------
template <bool HAS_BIAS, bool HAS_RES, bool DO_GELU, bool OUT_BF16>
__global__ __launch_bounds__(256)
void gemm_bf16_kernel(const unsigned short* __restrict__ A,
                      const unsigned short* __restrict__ Bt,
                      const float* __restrict__ bias,
                      const float* __restrict__ res,
                      float* __restrict__ outF,
                      unsigned short* __restrict__ outB,
                      int M, int N, int K) {
    __shared__ __attribute__((aligned(16))) unsigned short btile[2][64][32];
    const int lane = threadIdx.x & 31;
    const int wave = threadIdx.x >> 5;
    const int sel  = lane >> 4;
    const int l16  = lane & 15;
    const int n0   = blockIdx.x * 64;
    const int m0   = blockIdx.y * 256 + wave * 32;

    v8f zero = {};
    v8f acc[2][4];
#pragma unroll
    for (int ai = 0; ai < 2; ++ai)
#pragma unroll
        for (int j = 0; j < 4; ++j) acc[ai][j] = zero;

    gemm_core(A, Bt, K, m0, n0, acc, &btile[0][0][0]);

#pragma unroll
    for (int ai = 0; ai < 2; ++ai)
#pragma unroll
        for (int j = 0; j < 4; ++j)
#pragma unroll
            for (int r = 0; r < 8; ++r) {
                int m = m0 + ai * 16 + r + 8 * sel;
                int n = n0 + j * 16 + l16;
                float v = acc[ai][j][r];
                if (HAS_BIAS) v += bias[n];
                if (DO_GELU)  v = 0.5f * v * (1.0f + erff(v * 0.70710678118654752f));
                if (HAS_RES)  v += res[(size_t)m * N + n];
                if (OUT_BF16) outB[(size_t)m * N + n] = f2bf(v);
                else          outF[(size_t)m * N + n] = v;
            }
}

// ---------------- QKV GEMM (N=3072): writes Q,K row-major; V transposed [e][t] ----------------
__global__ __launch_bounds__(256)
void gemm_qkv_kernel(const unsigned short* __restrict__ A,
                     const unsigned short* __restrict__ Bt,
                     unsigned short* __restrict__ Q,
                     unsigned short* __restrict__ Kb,
                     unsigned short* __restrict__ Vt) {
    __shared__ __attribute__((aligned(16))) unsigned short btile[2][64][32];
    const int lane = threadIdx.x & 31;
    const int wave = threadIdx.x >> 5;
    const int sel  = lane >> 4;
    const int l16  = lane & 15;
    const int n0   = blockIdx.x * 64;          // which/head uniform per block
    const int m0   = blockIdx.y * 256 + wave * 32;

    v8f zero = {};
    v8f acc[2][4];
#pragma unroll
    for (int ai = 0; ai < 2; ++ai)
#pragma unroll
        for (int j = 0; j < 4; ++j) acc[ai][j] = zero;

    gemm_core(A, Bt, D_MODELc, m0, n0, acc, &btile[0][0][0]);

    const int which = n0 >> 10;              // 0=Q,1=K,2=V
    const int hh    = (n0 & 1023) >> 6;
#pragma unroll
    for (int ai = 0; ai < 2; ++ai)
#pragma unroll
        for (int j = 0; j < 4; ++j)
#pragma unroll
            for (int r = 0; r < 8; ++r) {
                int m = m0 + ai * 16 + r + 8 * sel;
                int b = m >> 11;             // row / S
                int s = m & (Sc - 1);
                int e = j * 16 + l16;
                unsigned short bv = f2bf(acc[ai][j][r]);
                size_t bh = (size_t)(b * N_HEADSc + hh);
                if (which == 0)      Q [(bh * Sc + s) * D_HEADc + e] = bv;
                else if (which == 1) Kb[(bh * Sc + s) * D_HEADc + e] = bv;
                else                 Vt[bh * D_HEADc * Sc + (size_t)e * Sc + s] = bv;
            }
}

// ---------------- flash attention: one 16-row Q tile per wave ----------------
__global__ __launch_bounds__(256)
void attention_kernel(const unsigned short* __restrict__ Q,
                      const unsigned short* __restrict__ Kb,
                      const unsigned short* __restrict__ Vt,
                      unsigned short* __restrict__ Ctx) {
    const int lane = threadIdx.x & 31;
    const int wave = threadIdx.x >> 5;
    const int sel  = lane >> 4;
    const int l16  = lane & 15;
    const int b    = blockIdx.z;
    const int h    = blockIdx.y;
    const int s0   = blockIdx.x * 128 + wave * 16;

    const size_t bh = (size_t)(b * N_HEADSc + h);
    const unsigned short* qb = Q  + bh * Sc * D_HEADc;
    const unsigned short* kb = Kb + bh * Sc * D_HEADc;
    const unsigned short* vb = Vt + bh * D_HEADc * Sc;

    __shared__ __attribute__((aligned(16))) unsigned short pbuf[8][16][32];

    FragAB qf0, qf1;
    const unsigned short* qrow = qb + (size_t)(s0 + l16) * D_HEADc;
    load_frag(qf0, qrow + sel * 8,      qrow + 16 + sel * 8);
    load_frag(qf1, qrow + 32 + sel * 8, qrow + 48 + sel * 8);

    v8f zero = {};
    v8f O[4];
#pragma unroll
    for (int j = 0; j < 4; ++j) O[j] = zero;
    float mi[8], li[8];
#pragma unroll
    for (int r = 0; r < 8; ++r) { mi[r] = -INFINITY; li[r] = 0.0f; }

    const float scale = 0.125f;   // 64^-0.5

    for (int tb = 0; tb < Sc; tb += 32) {
        v8f sc0 = zero, sc1 = zero;
        {
            FragAB kf0, kf1, kf2, kf3;
            const unsigned short* kr0 = kb + (size_t)(tb + l16) * D_HEADc;
            const unsigned short* kr1 = kb + (size_t)(tb + 16 + l16) * D_HEADc;
            load_frag(kf0, kr0 + sel * 8,      kr0 + 16 + sel * 8);
            load_frag(kf1, kr0 + 32 + sel * 8, kr0 + 48 + sel * 8);
            load_frag(kf2, kr1 + sel * 8,      kr1 + 16 + sel * 8);
            load_frag(kf3, kr1 + 32 + sel * 8, kr1 + 48 + sel * 8);
            sc0 = wmma_bf16(qf0.v, kf0.v, sc0);
            sc0 = wmma_bf16(qf1.v, kf1.v, sc0);
            sc1 = wmma_bf16(qf0.v, kf2.v, sc1);
            sc1 = wmma_bf16(qf1.v, kf3.v, sc1);
        }

        float p0[8], p1[8], alpha[8];
#pragma unroll
        for (int r = 0; r < 8; ++r) {
            float a = sc0[r] * scale;
            float c = sc1[r] * scale;
            float tm = fmaxf(a, c);
#pragma unroll
            for (int off = 1; off < 16; off <<= 1)
                tm = fmaxf(tm, __shfl_xor(tm, off, 32));
            float mn = fmaxf(mi[r], tm);
            alpha[r] = __expf(mi[r] - mn);
            p0[r] = __expf(a - mn);
            p1[r] = __expf(c - mn);
            float sm = p0[r] + p1[r];
#pragma unroll
            for (int off = 1; off < 16; off <<= 1)
                sm += __shfl_xor(sm, off, 32);
            li[r] = li[r] * alpha[r] + sm;
            mi[r] = mn;
        }
#pragma unroll
        for (int j = 0; j < 4; ++j)
#pragma unroll
            for (int r = 0; r < 8; ++r) O[j][r] *= alpha[r];

        // bounce P through per-wave LDS tile to convert C-layout -> A-layout
#pragma unroll
        for (int r = 0; r < 8; ++r) {
            int m = r + 8 * sel;
            pbuf[wave][m][l16]      = f2bf(p0[r]);
            pbuf[wave][m][16 + l16] = f2bf(p1[r]);
        }
        asm volatile("s_wait_dscnt 0" ::: "memory");   // DS in-order per wave; fence compiler too

        FragAB pf;
        const unsigned short* pr = &pbuf[wave][l16][0];
        load_frag(pf, pr + sel * 8, pr + 16 + sel * 8);

        FragAB vf0, vf1, vf2, vf3;
        const unsigned short* vr0 = vb + (size_t)(l16)      * Sc + tb;
        const unsigned short* vr1 = vb + (size_t)(16 + l16) * Sc + tb;
        const unsigned short* vr2 = vb + (size_t)(32 + l16) * Sc + tb;
        const unsigned short* vr3 = vb + (size_t)(48 + l16) * Sc + tb;
        load_frag(vf0, vr0 + sel * 8, vr0 + 16 + sel * 8);
        load_frag(vf1, vr1 + sel * 8, vr1 + 16 + sel * 8);
        load_frag(vf2, vr2 + sel * 8, vr2 + 16 + sel * 8);
        load_frag(vf3, vr3 + sel * 8, vr3 + 16 + sel * 8);
        O[0] = wmma_bf16(pf.v, vf0.v, O[0]);
        O[1] = wmma_bf16(pf.v, vf1.v, O[1]);
        O[2] = wmma_bf16(pf.v, vf2.v, O[2]);
        O[3] = wmma_bf16(pf.v, vf3.v, O[3]);
    }

    float inv[8];
#pragma unroll
    for (int r = 0; r < 8; ++r) inv[r] = 1.0f / li[r];
#pragma unroll
    for (int j = 0; j < 4; ++j)
#pragma unroll
        for (int r = 0; r < 8; ++r) {
            int m   = r + 8 * sel;
            int row = b * Sc + s0 + m;
            Ctx[(size_t)row * D_MODELc + h * D_HEADc + j * 16 + l16] =
                f2bf(O[j][r] * inv[r]);
        }
}

// ---------------- host orchestration ----------------
extern "C" void kernel_launch(void* const* d_in, const int* in_sizes, int n_in,
                              void* d_out, int out_size, void* d_ws, size_t ws_size,
                              hipStream_t stream) {
    (void)in_sizes; (void)n_in; (void)out_size; (void)ws_size;
    const float* x     = (const float*)d_in[0];
    const float* WQ    = (const float*)d_in[1];
    const float* WK    = (const float*)d_in[2];
    const float* WV    = (const float*)d_in[3];
    const float* WO    = (const float*)d_in[4];
    const float* ln1_g = (const float*)d_in[5];
    const float* ln1_b = (const float*)d_in[6];
    const float* ln2_g = (const float*)d_in[7];
    const float* ln2_b = (const float*)d_in[8];
    const float* W_in  = (const float*)d_in[9];
    const float* b_in  = (const float*)d_in[10];
    const float* W_out = (const float*)d_in[11];
    const float* b_out = (const float*)d_in[12];

    char* ws = (char*)d_ws;
    size_t off = 0;
    auto alloc = [&](size_t bytes) -> void* {
        void* p = ws + off;
        off += (bytes + 255) & ~(size_t)255;
        return p;
    };
    unsigned short* xln   = (unsigned short*)alloc((size_t)MROWS * D_MODELc * 2);
    unsigned short* wqkvt = (unsigned short*)alloc((size_t)3 * D_MODELc * D_MODELc * 2);
    unsigned short* wot   = (unsigned short*)alloc((size_t)D_MODELc * D_MODELc * 2);
    unsigned short* wint  = (unsigned short*)alloc((size_t)D_MLPc * D_MODELc * 2);
    unsigned short* woutt = (unsigned short*)alloc((size_t)D_MODELc * D_MLPc * 2);
    unsigned short* Qb    = (unsigned short*)alloc((size_t)MROWS * D_MODELc * 2);
    unsigned short* Kbf   = (unsigned short*)alloc((size_t)MROWS * D_MODELc * 2);
    unsigned short* Vtb   = (unsigned short*)alloc((size_t)MROWS * D_MODELc * 2);
    unsigned short* ctx   = (unsigned short*)alloc((size_t)MROWS * D_MODELc * 2);
    float*          x1    = (float*)alloc((size_t)MROWS * D_MODELc * 4);
    unsigned short* hm    = (unsigned short*)alloc((size_t)MROWS * D_MLPc * 2);

    dim3 blk(256);

    // Pack weights: Bt layout = [N][K] bf16
    transpose_cvt_kernel<<<dim3(256, 1, 16), blk, 0, stream>>>(
        WQ, wqkvt,                 1024, 64, 65536L, 65536L, 1024);
    transpose_cvt_kernel<<<dim3(256, 1, 16), blk, 0, stream>>>(
        WK, wqkvt + 1024 * 1024,   1024, 64, 65536L, 65536L, 1024);
    transpose_cvt_kernel<<<dim3(256, 1, 16), blk, 0, stream>>>(
        WV, wqkvt + 2048 * 1024,   1024, 64, 65536L, 65536L, 1024);
    transpose_cvt_kernel<<<dim3(256, 1, 16), blk, 0, stream>>>(
        WO, wot,                   64, 1024, 65536L, 64L,    1024);
    transpose_cvt_kernel<<<dim3(16384, 1, 1), blk, 0, stream>>>(
        W_in,  wint,  1024, 4096, 0L, 0L, 1024);
    transpose_cvt_kernel<<<dim3(16384, 1, 1), blk, 0, stream>>>(
        W_out, woutt, 4096, 1024, 0L, 0L, 4096);

    // LN1
    ln_bf16_kernel<<<MROWS, blk, 0, stream>>>(x, ln1_g, ln1_b, xln);

    // QKV
    gemm_qkv_kernel<<<dim3(3072 / 64, MROWS / 256), blk, 0, stream>>>(
        xln, wqkvt, Qb, Kbf, Vtb);

    // Attention (flash)
    attention_kernel<<<dim3(Sc / 128, N_HEADSc, Bc), blk, 0, stream>>>(
        Qb, Kbf, Vtb, ctx);

    // Output projection + residual -> x1 (f32)
    gemm_bf16_kernel<false, true, false, false>
        <<<dim3(1024 / 64, MROWS / 256), blk, 0, stream>>>(
            ctx, wot, nullptr, x, x1, nullptr, MROWS, D_MODELc, D_MODELc);

    // LN2 -> reuse xln
    ln_bf16_kernel<<<MROWS, blk, 0, stream>>>(x1, ln2_g, ln2_b, xln);

    // MLP in: +bias, exact GELU, bf16 out
    gemm_bf16_kernel<true, false, true, true>
        <<<dim3(4096 / 64, MROWS / 256), blk, 0, stream>>>(
            xln, wint, b_in, nullptr, nullptr, hm, MROWS, D_MLPc, D_MODELc);

    // MLP out: +bias, +residual x1 -> d_out (f32)
    gemm_bf16_kernel<true, true, false, false>
        <<<dim3(1024 / 64, MROWS / 256), blk, 0, stream>>>(
            hm, woutt, b_out, x1, (float*)d_out, nullptr, MROWS, D_MODELc, D_MLPc);
}
